// GAT_5806795784486
// MI455X (gfx1250) — matrix operand order repaired and checked by reading, not compiled, not run on hardware
//
#include <hip/hip_runtime.h>
#include <hip/hip_bf16.h>

#define DHEAD 64
#define GAT_SLOPE 0.2f

typedef __attribute__((ext_vector_type(16))) __bf16 v16bf;
typedef __attribute__((ext_vector_type(8)))  float  v8f;

// ---------- helpers ----------

// Order-preserving float -> uint mapping so segment-max can use integer atomics.
__device__ __forceinline__ unsigned f2ord(float f) {
    unsigned u = __float_as_uint(f);
    return (u & 0x80000000u) ? ~u : (u | 0x80000000u);
}
__device__ __forceinline__ float ord2f(unsigned o) {
    unsigned u = (o & 0x80000000u) ? (o & 0x7FFFFFFFu) : ~o;
    return __uint_as_float(u);
}

// ---------- weight pre-pack ----------
// f32 row-major W -> bf16 in WMMA B-fragment order over CTXP column tiles:
//   tiles [0, CT)  : W columns
//   tile  CT       : attention projections wl = W_h @ al_h (cols 0..H-1),
//                    wr = W_h @ ar_h (cols 8..8+H-1)  -> el/er fall out of the GEMM
//   tiles > CT     : zeros (padding so tiles distribute uniformly over waves)
// Element index: i = ((kb*CTXP + ct)*32 + lane)*16 + j, lane = hi*16 + m,
// holding B[K = kb*32 + (j < 8 ? hi*8 + j : 16 + hi*8 + (j-8))][col = ct*16 + m].
__global__ void pack_W_kernel(const float* __restrict__ W,
                              const float* __restrict__ al, const float* __restrict__ ar,
                              __bf16* __restrict__ Wf, int FIN, int FOUT, int HEADS, int CTXP)
{
    int CT = FOUT >> 4;
    int total = FIN * CTXP * 16;
    int i = blockIdx.x * blockDim.x + threadIdx.x;
    if (i >= total) return;
    int j    = i & 15;
    int l    = (i >> 4) & 31;
    int rest = i >> 9;
    int ct   = rest % CTXP;
    int kb   = rest / CTXP;
    int m = l & 15, hi = l >> 4;
    int K = kb * 32 + (j < 8 ? hi * 8 + j : 16 + hi * 8 + (j - 8));

    float val = 0.f;
    if (ct < CT) {
        val = W[(size_t)K * FOUT + ct * 16 + m];
    } else if (ct == CT) {
        if (m < HEADS) {                       // wl column for head m
            const float* wrow = W + (size_t)K * FOUT + m * DHEAD;
            const float* av   = al + m * DHEAD;
            for (int d = 0; d < DHEAD; ++d) val += wrow[d] * av[d];
        } else if (m >= 8 && m < 8 + HEADS) {  // wr column for head m-8
            int h = m - 8;
            const float* wrow = W + (size_t)K * FOUT + h * DHEAD;
            const float* av   = ar + h * DHEAD;
            for (int d = 0; d < DHEAD; ++d) val += wrow[d] * av[d];
        }
    }
    Wf[i] = (__bf16)val;
}

// ---------- fused feature transform + attention logits (pure WMMA) ----------
// 32 node rows per block; NW waves each own exactly TPW column tiles
// (NW*TPW = CT + 1 (+padding)), so the K loop is completely branchless.
// A rows are index-clamped (not predicated): tail lanes re-load the last valid
// row and their results are discarded by the store guards -> unconditional,
// fully pipelined A loads. B fragments are contiguous v16bf from the packed
// buffer. No LDS, no barriers.
template <int FIN, int FOUT, int HEADS, int NW, int TPW>
__global__ __launch_bounds__(NW * 32) void gemm_att_kernel(
    const float* __restrict__ in, const v16bf* __restrict__ Wf,
    float* __restrict__ hout, float* __restrict__ el, float* __restrict__ er,
    int N)
{
    constexpr int CT   = FOUT / 16;
    constexpr int CTXP = NW * TPW;
    static_assert(CTXP >= CT + 1, "need room for the el/er tile");
    constexpr int KB = FIN / 32;

    const int tid  = threadIdx.x;
    const int lane = tid & 31;
    const int m    = lane & 15;
    const int hi   = lane >> 4;
    const int row0 = blockIdx.x * 32;
    const int wu   = __builtin_amdgcn_readfirstlane(tid >> 5);  // uniform wave id

    v8f acc[2][TPW] = {};

    // clamp instead of predicate: out-of-range lanes duplicate the last row
    const int rowA0 = min(row0 + m,      N - 1);
    const int rowA1 = min(row0 + 16 + m, N - 1);
    const float* pA0 = in + (size_t)rowA0 * FIN + hi * 8;
    const float* pA1 = in + (size_t)rowA1 * FIN + hi * 8;

    for (int kb = 0; kb < KB; ++kb) {
        const int k0 = kb * 32;
        // A fragments: ISA 16-bit A 16x32 layout -> two contiguous 8-elem K runs/lane
        v16bf af[2];
#pragma unroll
        for (int rt = 0; rt < 2; ++rt) {
            const float* p = (rt ? pA1 : pA0) + k0;
            v8f a0 = *(const v8f*)p;         // K = hi*8 + 0..7
            v8f a1 = *(const v8f*)(p + 16);  // K = hi*8 + 16..23
#pragma unroll
            for (int j = 0; j < 8; ++j) {
                af[rt][j]     = (__bf16)a0[j];
                af[rt][j + 8] = (__bf16)a1[j];
            }
        }
#pragma unroll
        for (int t = 0; t < TPW; ++t) {
            const int ct = wu + t * NW;      // always valid: CTXP tiles exist
            v16bf bf = Wf[((size_t)kb * CTXP + ct) * 32 + lane];
            acc[0][t] = __builtin_amdgcn_wmma_f32_16x16x32_bf16(
                false, af[0], false, bf, (short)0, acc[0][t], false, false);
            acc[1][t] = __builtin_amdgcn_wmma_f32_16x16x32_bf16(
                false, af[1], false, bf, (short)0, acc[1][t], false, false);
        }
    }

    // epilogue: plain stores. Tiles < CT -> h; tile CT -> el/er; padding -> drop.
#pragma unroll
    for (int t = 0; t < TPW; ++t) {
        const int ct = wu + t * NW;
        if (ct < CT) {
            int col = ct * 16 + m;
#pragma unroll
            for (int rt = 0; rt < 2; ++rt) {
#pragma unroll
                for (int v = 0; v < 8; ++v) {
                    int row = row0 + rt * 16 + hi * 8 + v;
                    if (row < N) hout[(size_t)row * FOUT + col] = acc[rt][t][v];
                }
            }
        } else if (ct == CT) {
#pragma unroll
            for (int rt = 0; rt < 2; ++rt) {
#pragma unroll
                for (int v = 0; v < 8; ++v) {
                    int row = row0 + rt * 16 + hi * 8 + v;
                    float x = acc[rt][t][v];
                    if (row < N) {
                        if (m < HEADS)                    el[(size_t)row * HEADS + m] = x;
                        else if (m >= 8 && m < 8 + HEADS) er[(size_t)row * HEADS + (m - 8)] = x;
                    }
                }
            }
        }
    }
}

// ---------- edge softmax pipeline ----------

__global__ void fill_u32_kernel(unsigned* __restrict__ p, unsigned v, int n) {
    int i = blockIdx.x * blockDim.x + threadIdx.x;
    if (i < n) p[i] = v;
}

template <int HEADS>
__global__ void edge_logit_max_kernel(
    const int* __restrict__ src, const int* __restrict__ dst,
    const float* __restrict__ el, const float* __restrict__ er,
    float* __restrict__ ebuf, unsigned* __restrict__ mbuf, int E)
{
    int idx = blockIdx.x * blockDim.x + threadIdx.x;
    if (idx >= E * HEADS) return;
    int e = idx / HEADS, h = idx % HEADS;
    float v = el[src[e] * HEADS + h] + er[dst[e] * HEADS + h];
    v = (v > 0.f) ? v : GAT_SLOPE * v;           // leaky relu
    ebuf[idx] = v;
    atomicMax(&mbuf[dst[e] * HEADS + h], f2ord(v));
}

template <int HEADS>
__global__ void edge_exp_kernel(
    const int* __restrict__ dst, float* __restrict__ ebuf,
    const unsigned* __restrict__ mbuf, float* __restrict__ zbuf, int E)
{
    int idx = blockIdx.x * blockDim.x + threadIdx.x;
    if (idx >= E * HEADS) return;
    int e = idx / HEADS, h = idx % HEADS;
    int dh = dst[e] * HEADS + h;
    float ee = __expf(ebuf[idx] - ord2f(mbuf[dh]));
    ebuf[idx] = ee;
    unsafeAtomicAdd(&zbuf[dh], ee);
}

template <int HEADS>
__global__ void edge_alpha_kernel(
    const int* __restrict__ dst, float* __restrict__ ebuf,
    const float* __restrict__ zbuf, int E)
{
    int idx = blockIdx.x * blockDim.x + threadIdx.x;
    if (idx >= E * HEADS) return;
    int e = idx / HEADS, h = idx % HEADS;
    ebuf[idx] = ebuf[idx] / zbuf[dst[e] * HEADS + h];
}

// out[dst][c] += alpha[e][head(c)] * h[src][c] -- coalesced per-edge rows.
template <int FOUT, int HEADS>
__global__ void aggregate_kernel(
    const int* __restrict__ src, const int* __restrict__ dst,
    const float* __restrict__ ebuf, const float* __restrict__ hfeat,
    float* __restrict__ out, int E)
{
    unsigned idx = blockIdx.x * blockDim.x + threadIdx.x;
    unsigned total = (unsigned)E * (unsigned)FOUT;
    if (idx >= total) return;
    unsigned e = idx / FOUT;
    unsigned c = idx % FOUT;
    int h = (int)(c / DHEAD);
    float a = ebuf[e * HEADS + h];
    int s = src[e], dn = dst[e];
    unsafeAtomicAdd(&out[(size_t)dn * FOUT + c], a * hfeat[(size_t)s * FOUT + c]);
}

__global__ void bias_act_kernel(float* __restrict__ out, const float* __restrict__ b,
                                int total, int fmask, int do_elu)
{
    int idx = blockIdx.x * blockDim.x + threadIdx.x;
    if (idx >= total) return;
    float x = out[idx] + b[idx & fmask];
    if (do_elu) x = (x > 0.f) ? x : (__expf(x) - 1.f);
    out[idx] = x;
}

// ---------- launcher ----------

extern "C" void kernel_launch(void* const* d_in, const int* in_sizes, int n_in,
                              void* d_out, int out_size, void* d_ws, size_t ws_size,
                              hipStream_t stream)
{
    const float* x   = (const float*)d_in[0];
    const int*   src = (const int*)d_in[1];
    const int*   dst = (const int*)d_in[2];
    const float* W0  = (const float*)d_in[3];
    const float* al0 = (const float*)d_in[4];
    const float* ar0 = (const float*)d_in[5];
    const float* b0  = (const float*)d_in[6];
    const float* W1  = (const float*)d_in[7];
    const float* al1 = (const float*)d_in[8];
    const float* ar1 = (const float*)d_in[9];
    const float* b1  = (const float*)d_in[10];
    const float* W2  = (const float*)d_in[11];
    const float* al2 = (const float*)d_in[12];
    const float* ar2 = (const float*)d_in[13];
    const float* b2  = (const float*)d_in[14];
    float* out = (float*)d_out;

    const int N = in_sizes[0] / 128;
    const int E = in_sizes[1];

    // workspace layout
    float*    bufA = (float*)d_ws;                         // N*256 (transformed feats h)
    float*    bufB = bufA + (size_t)N * 256;               // N*256 (layer outputs)
    float*    el   = bufB + (size_t)N * 256;               // N*4
    float*    er   = el + (size_t)N * 4;                   // N*4
    unsigned* mb   = (unsigned*)(er + (size_t)N * 4);      // N*4
    float*    zb   = (float*)(mb + (size_t)N * 4);         // N*4
    float*    eb   = zb + (size_t)N * 4;                   // E*4
    __bf16*   Wf   = (__bf16*)(eb + (size_t)E * 4);        // <= 256*18*16 bf16 (144KB)

    const int TB = 256;
    const int gemm_blocks = (N + 31) / 32;
    const int eg4 = (E * 4 + 255) / 256;
    const int eg1 = (E + 255) / 256;

    // ======== layer 0: 128 -> 4x64, ELU ======== (18 tiles, 9 waves x 2)
    pack_W_kernel<<<(128 * 18 * 16 + 255) / 256, TB, 0, stream>>>(W0, al0, ar0, Wf, 128, 256, 4, 18);
    gemm_att_kernel<128, 256, 4, 9, 2><<<gemm_blocks, 288, 0, stream>>>(
        x, (const v16bf*)Wf, bufA, el, er, N);
    fill_u32_kernel<<<(N * 4 + 255) / 256, TB, 0, stream>>>(mb, 0u, N * 4);
    fill_u32_kernel<<<(N * 4 + 255) / 256, TB, 0, stream>>>((unsigned*)zb, 0u, N * 4);
    fill_u32_kernel<<<(N * 256 + 255) / 256, TB, 0, stream>>>((unsigned*)bufB, 0u, N * 256);
    edge_logit_max_kernel<4><<<eg4, TB, 0, stream>>>(src, dst, el, er, eb, mb, E);
    edge_exp_kernel<4><<<eg4, TB, 0, stream>>>(dst, eb, mb, zb, E);
    edge_alpha_kernel<4><<<eg4, TB, 0, stream>>>(dst, eb, zb, E);
    {
        unsigned total = (unsigned)E * 256u;
        aggregate_kernel<256, 4><<<(total + 255) / 256, TB, 0, stream>>>(src, dst, eb, bufA, bufB, E);
    }
    bias_act_kernel<<<(N * 256 + 255) / 256, TB, 0, stream>>>(bufB, b0, N * 256, 255, 1);

    // ======== layer 1: 256 -> 4x64, ELU ======== (18 tiles, 9 waves x 2)
    pack_W_kernel<<<(256 * 18 * 16 + 255) / 256, TB, 0, stream>>>(W1, al1, ar1, Wf, 256, 256, 4, 18);
    gemm_att_kernel<256, 256, 4, 9, 2><<<gemm_blocks, 288, 0, stream>>>(
        bufB, (const v16bf*)Wf, bufA, el, er, N);
    fill_u32_kernel<<<(N * 4 + 255) / 256, TB, 0, stream>>>(mb, 0u, N * 4);
    fill_u32_kernel<<<(N * 4 + 255) / 256, TB, 0, stream>>>((unsigned*)zb, 0u, N * 4);
    fill_u32_kernel<<<(N * 256 + 255) / 256, TB, 0, stream>>>((unsigned*)bufB, 0u, N * 256); // after gemm
    edge_logit_max_kernel<4><<<eg4, TB, 0, stream>>>(src, dst, el, er, eb, mb, E);
    edge_exp_kernel<4><<<eg4, TB, 0, stream>>>(dst, eb, mb, zb, E);
    edge_alpha_kernel<4><<<eg4, TB, 0, stream>>>(dst, eb, zb, E);
    {
        unsigned total = (unsigned)E * 256u;
        aggregate_kernel<256, 4><<<(total + 255) / 256, TB, 0, stream>>>(src, dst, eb, bufA, bufB, E);
    }
    bias_act_kernel<<<(N * 256 + 255) / 256, TB, 0, stream>>>(bufB, b1, N * 256, 255, 1);

    // ======== layer 2: 256 -> 1x64, no act ======== (5 tiles, 5 waves x 1)
    pack_W_kernel<<<(256 * 5 * 16 + 255) / 256, TB, 0, stream>>>(W2, al2, ar2, Wf, 256, 64, 1, 5);
    gemm_att_kernel<256, 64, 1, 5, 1><<<gemm_blocks, 160, 0, stream>>>(
        bufB, (const v16bf*)Wf, bufA, el, er, N);
    fill_u32_kernel<<<(N + 255) / 256, TB, 0, stream>>>(mb, 0u, N);
    fill_u32_kernel<<<(N + 255) / 256, TB, 0, stream>>>((unsigned*)zb, 0u, N);
    fill_u32_kernel<<<(N * 64 + 255) / 256, TB, 0, stream>>>((unsigned*)out, 0u, N * 64);
    edge_logit_max_kernel<1><<<eg1, TB, 0, stream>>>(src, dst, el, er, eb, mb, E);
    edge_exp_kernel<1><<<eg1, TB, 0, stream>>>(dst, eb, mb, zb, E);
    edge_alpha_kernel<1><<<eg1, TB, 0, stream>>>(dst, eb, zb, E);
    {
        unsigned total = (unsigned)E * 64u;
        aggregate_kernel<64, 1><<<(total + 255) / 256, TB, 0, stream>>>(src, dst, eb, bufA, out, E);
    }
    bias_act_kernel<<<(N * 64 + 255) / 256, TB, 0, stream>>>(out, b2, N * 64, 63, 0);
}
